// TGNNModel_70574902608402
// MI455X (gfx1250) — matrix operand order repaired and checked by reading, not compiled, not run on hardware
//
#include <hip/hip_runtime.h>
#include <hip/hip_bf16.h>
#include <math.h>

typedef __attribute__((ext_vector_type(16))) _Float16 v16h;
typedef __attribute__((ext_vector_type(4)))  _Float16 v4h;
typedef __attribute__((ext_vector_type(8)))  float    v8f;

#define DHID 128
#define NEG_SLOPE 0.2f
#define LN_EPS 1e-5f

// ---------------------------------------------------------------------------
// float atomic max via sign-aware integer punning (init with -inf bits)
// ---------------------------------------------------------------------------
__device__ __forceinline__ void atomicMaxF(float* addr, float v) {
    if (v >= 0.f) atomicMax((int*)addr, __float_as_int(v));
    else          atomicMin((unsigned int*)addr, __float_as_uint(v));
}

// ---------------------------------------------------------------------------
// Convert f32 weight [K x Nc] -> f16 transposed [Nc x Kpad], zero-padded K
// ---------------------------------------------------------------------------
__global__ void k_convert_wt(const float* __restrict__ W, _Float16* __restrict__ Wt,
                             int K, int Nc, int Kpad) {
    int idx = blockIdx.x * blockDim.x + threadIdx.x;
    if (idx >= Nc * Kpad) return;
    int n = idx / Kpad, k = idx - n * Kpad;
    Wt[idx] = (k < K) ? (_Float16)W[(size_t)k * Nc + n] : (_Float16)0.f;
}

__device__ __forceinline__ v8f wmma_f16(const v16h& a, const v16h& b, const v8f& c) {
    return __builtin_amdgcn_wmma_f32_16x16x32_f16(false, a, false, b,
                                                  (short)0, c, false, false);
}

// ---------------------------------------------------------------------------
// GEMM: C[M x 128] = A[M x K (lda)] @ W[K x 128] (+bias).
// A: f32 row-major, W: f16 transposed [128 x Kpad].
// Block tile: 32 rows x 128 cols; 8 waves, each wave = 16 cols x 2 row tiles
// (2 WMMAs share one B fragment). LDS A tile is f16, double-buffered ->
// one barrier per 32-wide k-step. Steady-state A staging is one b128 load
// per thread; bounds checks only in the uniform tail step.
// ---------------------------------------------------------------------------
__global__ __launch_bounds__(256)
void k_gemm(const float* __restrict__ A, int lda, int K, int M,
            const _Float16* __restrict__ Bt, int Kpad,
            const float* __restrict__ bias, float* __restrict__ C) {
    __shared__ _Float16 As[2][32][32];
    const int t    = threadIdx.x;
    const int lane = t & 31;
    const int wave = t >> 5;
    const int half = lane >> 4;        // 0: lanes 0-15, 1: lanes 16-31
    const int ar   = lane & 15;
    const int kb   = half * 8;         // A fragment K base (ISA 16-bit A layout)
    const int kb2  = half * 8;         // B fragment K base
    const int bc   = wave * 16 + ar;   // global output column (0..127)
    const int row0 = blockIdx.x * 32;

    // A staging: thread loads 4 consecutive floats of one of the 32 rows
    const int sr = t >> 3;             // 0..31
    const int sc = (t & 7) * 4;        // 0,4,...,28
    int gr = row0 + sr; if (gr >= M) gr = M - 1;   // clamp: no divergent loads
    const float* Arow = A + (size_t)gr * lda;

    auto load_tile = [&](int k0) -> float4 {
        float4 fv;
        int gk = k0 + sc;
        if (k0 + 32 <= K) {            // uniform (scalar) condition
            fv = *(const float4*)(Arow + gk);
        } else {
            fv.x = (gk + 0 < K) ? Arow[gk + 0] : 0.f;
            fv.y = (gk + 1 < K) ? Arow[gk + 1] : 0.f;
            fv.z = (gk + 2 < K) ? Arow[gk + 2] : 0.f;
            fv.w = (gk + 3 < K) ? Arow[gk + 3] : 0.f;
        }
        return fv;
    };
    auto store_tile = [&](int buf, const float4& fv) {
        v4h hv;
        hv[0] = (_Float16)fv.x; hv[1] = (_Float16)fv.y;
        hv[2] = (_Float16)fv.z; hv[3] = (_Float16)fv.w;
        *(v4h*)&As[buf][sr][sc] = hv;  // ds_store_b64
    };

    const int ksteps = Kpad >> 5;
    float4 fv = load_tile(0);
    store_tile(0, fv);
    __syncthreads();

    v8f acc0 = {}, acc1 = {};
    for (int ks = 0; ks < ksteps; ++ks) {
        const int buf = ks & 1;
        const bool has_next = (ks + 1 < ksteps);
        float4 nxt;
        if (has_next) nxt = load_tile((ks + 1) << 5);   // overlap with WMMA

        v16h a0, a1, b;
#pragma unroll
        for (int i = 0; i < 8; ++i) {
            a0[i]     = As[buf][ar][kb + i];
            a0[i + 8] = As[buf][ar][16 + kb + i];
            a1[i]     = As[buf][16 + ar][kb + i];
            a1[i + 8] = As[buf][16 + ar][16 + kb + i];
        }
        const _Float16* bp = Bt + (size_t)bc * Kpad + (ks << 5) + kb2;
#pragma unroll
        for (int i = 0; i < 8; ++i) { b[i] = bp[i]; b[i + 8] = bp[16 + i]; }

        acc0 = wmma_f16(a0, b, acc0);
        acc1 = wmma_f16(a1, b, acc1);

        if (has_next) {
            store_tile((ks + 1) & 1, nxt);
            __syncthreads();           // single barrier per k-step
        }
    }

    const float bv = bias ? bias[bc] : 0.f;
    const int rb0 = row0 + half * 8;
#pragma unroll
    for (int i = 0; i < 8; ++i) {
        int r0 = rb0 + i, r1 = rb0 + 16 + i;
        if (r0 < M) C[(size_t)r0 * DHID + bc] = acc0[i] + bv;
        if (r1 < M) C[(size_t)r1 * DHID + bc] = acc1[i] + bv;
    }
}

// ---------------------------------------------------------------------------
// LayerNorm over 128 features; one wave per node, 4 elems per lane.
// out = LN(in1 (+in2) (+addb)) * g + b
// ---------------------------------------------------------------------------
__global__ __launch_bounds__(256)
void k_ln(const float* __restrict__ in1, const float* __restrict__ in2,
          const float* __restrict__ addb,
          const float* __restrict__ g, const float* __restrict__ bta,
          float* __restrict__ out, int n) {
    int node = (blockIdx.x * blockDim.x + threadIdx.x) >> 5;
    int lane = threadIdx.x & 31;
    if (node >= n) return;
    float4 p1 = *(const float4*)(in1 + (size_t)node * DHID + lane * 4);
    float y[4] = {p1.x, p1.y, p1.z, p1.w};
    if (in2) {
        float4 p2 = *(const float4*)(in2 + (size_t)node * DHID + lane * 4);
        y[0] += p2.x; y[1] += p2.y; y[2] += p2.z; y[3] += p2.w;
    }
    if (addb) {
        float4 pb = *(const float4*)(addb + lane * 4);
        y[0] += pb.x; y[1] += pb.y; y[2] += pb.z; y[3] += pb.w;
    }
    float s = y[0] + y[1] + y[2] + y[3];
#pragma unroll
    for (int mk = 1; mk < 32; mk <<= 1) s += __shfl_xor(s, mk, 32);
    float mu = s * (1.f / DHID);
    float d[4], vs = 0.f;
#pragma unroll
    for (int j = 0; j < 4; ++j) { d[j] = y[j] - mu; vs += d[j] * d[j]; }
#pragma unroll
    for (int mk = 1; mk < 32; mk <<= 1) vs += __shfl_xor(vs, mk, 32);
    float rstd = rsqrtf(vs * (1.f / DHID) + LN_EPS);
    float4 gv = *(const float4*)(g   + lane * 4);
    float4 bv = *(const float4*)(bta + lane * 4);
    float4 o;
    o.x = d[0] * rstd * gv.x + bv.x;
    o.y = d[1] * rstd * gv.y + bv.y;
    o.z = d[2] * rstd * gv.z + bv.z;
    o.w = d[3] * rstd * gv.w + bv.w;
    *(float4*)(out + (size_t)node * DHID + lane * 4) = o;
}

// ---------------------------------------------------------------------------
// Attention logits: a_s[n,h] = <hW[n,h,:], att_src[h,:]>, likewise a_d.
// ---------------------------------------------------------------------------
__global__ void k_att(const float* __restrict__ hw,
                      const float* __restrict__ ws, const float* __restrict__ wd,
                      float* __restrict__ a_s, float* __restrict__ a_d, int n) {
    int idx = blockIdx.x * blockDim.x + threadIdx.x;
    if (idx >= n * 4) return;
    int node = idx >> 2, hh = idx & 3;
    const float* hp = hw + (size_t)node * DHID + hh * 32;
    const float* sp = ws + hh * 32;
    const float* dp = wd + hh * 32;
    float ss = 0.f, sd = 0.f;
#pragma unroll 8
    for (int c = 0; c < 32; ++c) { float v = hp[c]; ss += v * sp[c]; sd += v * dp[c]; }
    a_s[idx] = ss; a_d[idx] = sd;
}

// init m=-inf, z=0, agg=0
__global__ void k_init(float* __restrict__ agg, float* __restrict__ m,
                       float* __restrict__ z, int n) {
    size_t idx = (size_t)blockIdx.x * blockDim.x + threadIdx.x;
    if (idx < (size_t)n * DHID) agg[idx] = 0.f;
    if (idx < (size_t)n * 4) { m[idx] = -INFINITY; z[idx] = 0.f; }
}

__device__ __forceinline__ void edge_sd(const int* src, const int* dst, int E,
                                        int idx, int& s, int& d) {
    if (idx < E) { s = src[idx]; d = dst[idx]; } else { s = d = idx - E; }
}

// pass 1: per-destination, per-head max of leaky_relu(a_s[s]+a_d[d])
__global__ void k_edge_max(const int* __restrict__ src, const int* __restrict__ dst,
                           int E, int EN, const float* __restrict__ a_s,
                           const float* __restrict__ a_d, float* __restrict__ m) {
    int idx = blockIdx.x * blockDim.x + threadIdx.x;
    if (idx >= EN) return;
    int s, d; edge_sd(src, dst, E, idx, s, d);
    float4 as4 = *(const float4*)(a_s + (size_t)s * 4);
    float4 ad4 = *(const float4*)(a_d + (size_t)d * 4);
    float e[4] = {as4.x + ad4.x, as4.y + ad4.y, as4.z + ad4.z, as4.w + ad4.w};
#pragma unroll
    for (int hh = 0; hh < 4; ++hh) {
        float ev = e[hh] > 0.f ? e[hh] : NEG_SLOPE * e[hh];
        atomicMaxF(&m[(size_t)d * 4 + hh], ev);
    }
}

// pass 2: z[d,h] += exp(e - m[d,h])
__global__ void k_edge_sum(const int* __restrict__ src, const int* __restrict__ dst,
                           int E, int EN, const float* __restrict__ a_s,
                           const float* __restrict__ a_d, const float* __restrict__ m,
                           float* __restrict__ z) {
    int idx = blockIdx.x * blockDim.x + threadIdx.x;
    if (idx >= EN) return;
    int s, d; edge_sd(src, dst, E, idx, s, d);
    float4 as4 = *(const float4*)(a_s + (size_t)s * 4);
    float4 ad4 = *(const float4*)(a_d + (size_t)d * 4);
    float4 m4  = *(const float4*)(m   + (size_t)d * 4);
    float e[4] = {as4.x + ad4.x, as4.y + ad4.y, as4.z + ad4.z, as4.w + ad4.w};
    float mm[4] = {m4.x, m4.y, m4.z, m4.w};
#pragma unroll
    for (int hh = 0; hh < 4; ++hh) {
        float ev = e[hh] > 0.f ? e[hh] : NEG_SLOPE * e[hh];
        atomicAdd(&z[(size_t)d * 4 + hh], expf(ev - mm[hh]));
    }
}

// pass 3: agg[d,:] += hW[s,:] * alpha   (one wave per edge, float4 per lane)
__global__ __launch_bounds__(256)
void k_edge_agg(const int* __restrict__ src, const int* __restrict__ dst,
                int E, int EN, const float* __restrict__ a_s,
                const float* __restrict__ a_d, const float* __restrict__ m,
                const float* __restrict__ z, const float* __restrict__ hw,
                float* __restrict__ agg) {
    int gid  = blockIdx.x * blockDim.x + threadIdx.x;
    int edge = gid >> 5;
    if (edge >= EN) return;
    int lane = gid & 31;
    int s, d; edge_sd(src, dst, E, edge, s, d);
    int hh = lane >> 3;   // head for channels [4*lane, 4*lane+4)
    float e = a_s[(size_t)s * 4 + hh] + a_d[(size_t)d * 4 + hh];
    e = e > 0.f ? e : NEG_SLOPE * e;
    float alpha = expf(e - m[(size_t)d * 4 + hh]) / z[(size_t)d * 4 + hh];
    float4 hv = *(const float4*)(hw + (size_t)s * DHID + lane * 4);
    float* op = agg + (size_t)d * DHID + lane * 4;
    atomicAdd(op + 0, hv.x * alpha);
    atomicAdd(op + 1, hv.y * alpha);
    atomicAdd(op + 2, hv.z * alpha);
    atomicAdd(op + 3, hv.w * alpha);
}

// ---------------------------------------------------------------------------
// Fused head: per block of 16 edges, ef=[h[src],h[dst]] (16x256) -> f16 LDS,
// hid = relu(ef@W1+b1) via WMMA (8 waves x 16 cols), out = hid@W2 + b2.
// ---------------------------------------------------------------------------
__global__ __launch_bounds__(256)
void k_head(const float* __restrict__ h, const int* __restrict__ src,
            const int* __restrict__ dst, const _Float16* __restrict__ W1t,
            const float* __restrict__ b1, const float* __restrict__ W2,
            const float* __restrict__ b2, float* __restrict__ out, int E) {
    __shared__ _Float16 Ef[16][256];
    __shared__ float so[16];
    const int t  = threadIdx.x;
    const int e0 = blockIdx.x * 16;
    {
        int r  = t >> 4;
        int cg = (t & 15) << 4;
        int e  = e0 + r; if (e >= E) e = E - 1;      // clamp keeps EXEC full
        int s = src[e], d = dst[e];
        const float* p = (cg < DHID) ? (h + (size_t)s * DHID + cg)
                                     : (h + (size_t)d * DHID + (cg - DHID));
#pragma unroll
        for (int j = 0; j < 16; ++j) Ef[r][cg + j] = (_Float16)p[j];
    }
    if (t < 16) so[t] = 0.f;
    __syncthreads();

    const int lane = t & 31, wave = t >> 5;
    const int half = lane >> 4, ar = lane & 15;
    const int kb = half * 8, kb2 = half * 8;
    const int bc = wave * 16 + ar;
    v8f acc = {};
#pragma unroll
    for (int ks = 0; ks < 8; ++ks) {
        const int k0 = ks << 5;
        v16h a, b;
#pragma unroll
        for (int i = 0; i < 8; ++i) {
            a[i]     = Ef[ar][k0 + kb + i];
            a[i + 8] = Ef[ar][k0 + 16 + kb + i];
        }
        const _Float16* bp = W1t + (size_t)bc * 256 + k0 + kb2;
#pragma unroll
        for (int i = 0; i < 8; ++i) { b[i] = bp[i]; b[i + 8] = bp[16 + i]; }
        acc = wmma_f16(a, b, acc);
    }
    float b1v = b1[bc], w2v = W2[bc];
    float p8[8];
#pragma unroll
    for (int i = 0; i < 8; ++i) {
        float hv = acc[i] + b1v;
        p8[i] = fmaxf(hv, 0.f) * w2v;
    }
#pragma unroll
    for (int mk = 1; mk < 16; mk <<= 1) {
#pragma unroll
        for (int i = 0; i < 8; ++i) p8[i] += __shfl_xor(p8[i], mk, 32);
    }
    if (ar == 0) {
#pragma unroll
        for (int i = 0; i < 8; ++i) atomicAdd(&so[half * 8 + i], p8[i]);
    }
    __syncthreads();
    if (t < 16 && (e0 + t) < E) out[e0 + t] = so[t] + b2[0];
}

// ---------------------------------------------------------------------------
extern "C" void kernel_launch(void* const* d_in, const int* in_sizes, int n_in,
                              void* d_out, int out_size, void* d_ws, size_t ws_size,
                              hipStream_t stream) {
    const int Nn = in_sizes[0] / 776;
    const int Ee = in_sizes[1] / 2;
    const int EN = Ee + Nn;

    const float* x   = (const float*)d_in[0];
    const int*   src = (const int*)d_in[1];
    const int*   dst = src + Ee;
    const float* Wp  = (const float*)d_in[3];
    const float* bp  = (const float*)d_in[4];
    const float* W1  = (const float*)d_in[5];
    const float* b1  = (const float*)d_in[6];
    const float* W2  = (const float*)d_in[7];
    const float* b2  = (const float*)d_in[8];

    char* w = (char*)d_ws;
    float* hbuf = (float*)w; w += (size_t)Nn * DHID * 4;
    float* xt   = (float*)w; w += (size_t)Nn * DHID * 4;
    float* hw   = (float*)w; w += (size_t)Nn * DHID * 4;
    float* agg  = (float*)w; w += (size_t)Nn * DHID * 4;
    float* a_s  = (float*)w; w += (size_t)Nn * 4 * 4;
    float* a_d  = (float*)w; w += (size_t)Nn * 4 * 4;
    float* mbuf = (float*)w; w += (size_t)Nn * 4 * 4;
    float* zbuf = (float*)w; w += (size_t)Nn * 4 * 4;
    _Float16* wt = (_Float16*)w;  // reused: max 128*800 halves

    const int mtiles = (Nn + 31) / 32;

    // input projection: h = x @ Wp + bp   (K=776 padded to 800)
    k_convert_wt<<<(DHID * 800 + 255) / 256, 256, 0, stream>>>(Wp, wt, 776, DHID, 800);
    k_gemm<<<mtiles, 256, 0, stream>>>(x, 776, 776, Nn, wt, 800, bp, hbuf);

    for (int L = 0; L < 2; ++L) {
        const int base = 9 + L * 8;
        const float* ln_a_g = (const float*)d_in[base + 0];
        const float* ln_a_b = (const float*)d_in[base + 1];
        const float* gat_W  = (const float*)d_in[base + 2];
        const float* gat_as = (const float*)d_in[base + 3];
        const float* gat_ad = (const float*)d_in[base + 4];
        const float* gat_b  = (const float*)d_in[base + 5];
        const float* ln_b_g = (const float*)d_in[base + 6];
        const float* ln_b_b = (const float*)d_in[base + 7];

        k_ln<<<(Nn * 32 + 255) / 256, 256, 0, stream>>>(
            hbuf, nullptr, nullptr, ln_a_g, ln_a_b, xt, Nn);
        k_convert_wt<<<(DHID * DHID + 255) / 256, 256, 0, stream>>>(
            gat_W, wt, DHID, DHID, DHID);
        k_gemm<<<mtiles, 256, 0, stream>>>(xt, DHID, DHID, Nn, wt, DHID, nullptr, hw);
        k_att<<<(Nn * 4 + 255) / 256, 256, 0, stream>>>(hw, gat_as, gat_ad, a_s, a_d, Nn);
        k_init<<<(int)(((size_t)Nn * DHID + 255) / 256), 256, 0, stream>>>(
            agg, mbuf, zbuf, Nn);
        k_edge_max<<<(EN + 255) / 256, 256, 0, stream>>>(src, dst, Ee, EN, a_s, a_d, mbuf);
        k_edge_sum<<<(EN + 255) / 256, 256, 0, stream>>>(src, dst, Ee, EN, a_s, a_d, mbuf, zbuf);
        k_edge_agg<<<(int)(((size_t)EN * 32 + 255) / 256), 256, 0, stream>>>(
            src, dst, Ee, EN, a_s, a_d, mbuf, zbuf, hw, agg);
        k_ln<<<(Nn * 32 + 255) / 256, 256, 0, stream>>>(
            xt, agg, gat_b, ln_b_g, ln_b_b, hbuf, Nn);
    }

    // diffusion head: out = relu([h[src],h[dst]] @ W1 + b1) @ W2 + b2
    k_convert_wt<<<(DHID * 256 + 255) / 256, 256, 0, stream>>>(W1, wt, 256, DHID, 256);
    k_head<<<(Ee + 15) / 16, 256, 0, stream>>>(hbuf, src, dst, wt, b1, W2, b2,
                                               (float*)d_out, Ee);
    (void)n_in; (void)out_size; (void)ws_size;
}
